// SelfAttention_43894565765797
// MI455X (gfx1250) — matrix operand order repaired and checked by reading, not compiled, not run on hardware
//
#include <hip/hip_runtime.h>
#include <hip/hip_bf16.h>
#include <math.h>

// ---------------------------------------------------------------------------
// Self-attention forward for MI455X (gfx1250, wave32, WMMA).
//   B=4, L=2048, D=1024, H=16, DH=64
//   proj_kernel: [Q|K|V] = x @ [Wq|Wkv]  (f32 in, f16 out, WMMA f16->f32)
//   attn_kernel: per-wave 16-row flash attention; per-block K/V tiles staged
//                to LDS with double-buffered global_load_async_to_lds_b128.
// ---------------------------------------------------------------------------

typedef __attribute__((ext_vector_type(16))) _Float16 v16h;
typedef __attribute__((ext_vector_type(8)))  float    v8f;
typedef __attribute__((ext_vector_type(4)))  unsigned int v4u;

#define B_  4
#define L_  2048
#define D_  1024
#define H_  16
#define DH_ 64
#define QSCALE 0.125f   // DH^-0.5

union FragU { v16h h; v4u u[2]; };

// A-matrix 16x32 f16 fragment (ISA 7.12.2): lane l holds row m=l%16,
// halves 0..7  = K in [8h, 8h+7]        -> 16B at byte offset 16*h
// halves 8..15 = K in [16+8h, 16+8h+7]  -> 16B at byte offset 32+16*h
__device__ __forceinline__ v16h frag_a_f16(const _Float16* rowp, int half) {
  FragU f;
  const char* p = (const char*)rowp;
  f.u[0] = *(const v4u*)(p + 16 * half);
  f.u[1] = *(const v4u*)(p + 32 + 16 * half);
  return f.h;
}

// B-matrix 32x16 f16 fragment: lane l holds column n=l%16 of B,
// halves 0..7 = K in [8h, 8h+7], halves 8..15 = K in [16+8h, 16+8h+7] of row n of B^T.
__device__ __forceinline__ v16h frag_b_f16(const _Float16* rowTp, int half) {
  FragU f;
  const char* p = (const char*)rowTp;
  f.u[0] = *(const v4u*)(p + 16 * half);
  f.u[1] = *(const v4u*)(p + 32 + 16 * half);
  return f.h;
}

// A fragment built from an f32 row (convert on the fly; lowers to v_cvt_pk).
__device__ __forceinline__ v16h frag_a_f32(const float* rowp, int half) {
  v16h o;
  const float* p0 = rowp + 8 * half;
  const float* p1 = rowp + 16 + 8 * half;
#pragma unroll
  for (int i = 0; i < 8; ++i) {
    o[i]     = (_Float16)p0[i];
    o[8 + i] = (_Float16)p1[i];
  }
  return o;
}

__device__ __forceinline__ v8f wmma16(v16h a, v16h b, v8f c) {
  return __builtin_amdgcn_wmma_f32_16x16x32_f16(
      false, a, false, b, (short)0, c, false, false);
}

__device__ __forceinline__ v8f v8f_zero() {
  v8f z;
#pragma unroll
  for (int i = 0; i < 8; ++i) z[i] = 0.0f;
  return z;
}

// Async 16B copy global -> LDS (ASYNCcnt-tracked, gfx1250).
__device__ __forceinline__ void async_cp_b128(void* lds_dst, const _Float16* gsrc) {
  unsigned off = (unsigned)(uintptr_t)lds_dst;   // low 32 bits = LDS offset
  asm volatile("global_load_async_to_lds_b128 %0, %1, off"
               :: "v"(off), "v"(gsrc) : "memory");
}
__device__ __forceinline__ void wait_async0() {
  asm volatile("s_wait_asynccnt 0" ::: "memory");
}

// ---------------------------------------------------------------------------
// Kernel 1: fused projection GEMM.
// grid = (64 row-blocks of 128, 48 col-blocks of 64), block = 256 (8 waves).
// ---------------------------------------------------------------------------
__global__ __launch_bounds__(256) void proj_kernel(
    const float* __restrict__ x, const float* __restrict__ Wq,
    const float* __restrict__ Wkv, _Float16* __restrict__ Qh,
    _Float16* __restrict__ Kh, _Float16* __restrict__ Vt) {
  __shared__ alignas(16) _Float16 wt[64][40];  // W^T tile [col][k], padded

  const int tid  = threadIdx.x;
  const int wave = tid >> 5;
  const int lane = tid & 31;
  const int m16  = lane & 15;
  const int half = lane >> 4;
  const int row0 = blockIdx.x * 128 + wave * 16;
  const int colbase = blockIdx.y * 64;

  const float* wsrc;
  int wstride, wcol0;
  if (colbase < D_) { wsrc = Wq;  wstride = D_;     wcol0 = colbase; }
  else              { wsrc = Wkv; wstride = 2 * D_; wcol0 = colbase - D_; }

  v8f acc[4];
#pragma unroll
  for (int j = 0; j < 4; ++j) acc[j] = v8f_zero();

  for (int kb = 0; kb < D_; kb += 32) {
    __syncthreads();
#pragma unroll
    for (int i = 0; i < 8; ++i) {
      int idx = tid + i * 256;   // 0..2047 = 32k x 64c
      int k = idx >> 6;
      int c = idx & 63;
      wt[c][k] = (_Float16)wsrc[(size_t)(kb + k) * wstride + wcol0 + c];
    }
    __syncthreads();

    v16h a = frag_a_f32(x + (size_t)(row0 + m16) * D_ + kb, half);
#pragma unroll
    for (int j = 0; j < 4; ++j) {
      v16h b = frag_b_f16(&wt[j * 16 + m16][0], half);
      acc[j] = wmma16(a, b, acc[j]);
    }
  }

  // Epilogue: Q/K/V choice and head index are uniform per block (64-col
  // blocks never straddle the 1024-wide regions) -> no per-element branches.
  if (colbase < D_) {
    const int h = colbase >> 6;
#pragma unroll
    for (int j = 0; j < 4; ++j) {
      const int dh = j * 16 + m16;
#pragma unroll
      for (int r = 0; r < 8; ++r) {
        int grow = row0 + r + 8 * half;
        int bb = grow >> 11, l = grow & (L_ - 1);
        Qh[(((size_t)(bb * H_ + h)) * L_ + l) * DH_ + dh] =
            (_Float16)(acc[j][r] * QSCALE);
      }
    }
  } else if (colbase < 2 * D_) {
    const int h = (colbase - D_) >> 6;
#pragma unroll
    for (int j = 0; j < 4; ++j) {
      const int dh = j * 16 + m16;
#pragma unroll
      for (int r = 0; r < 8; ++r) {
        int grow = row0 + r + 8 * half;
        int bb = grow >> 11, l = grow & (L_ - 1);
        Kh[(((size_t)(bb * H_ + h)) * L_ + l) * DH_ + dh] = (_Float16)acc[j][r];
      }
    }
  } else {
    const int h = (colbase - 2 * D_) >> 6;
#pragma unroll
    for (int j = 0; j < 4; ++j) {
      const int dh = j * 16 + m16;
#pragma unroll
      for (int r = 0; r < 8; ++r) {
        int grow = row0 + r + 8 * half;
        int bb = grow >> 11, l = grow & (L_ - 1);
        Vt[(((size_t)(bb * H_ + h)) * DH_ + dh) * L_ + l] = (_Float16)acc[j][r];
      }
    }
  }
}

// ---------------------------------------------------------------------------
// Kernel 2: flash attention.
// Block = 256 threads (8 waves) = 8 consecutive q-tiles of ONE (b,h), so the
// K/V tiles for each 32-key step are shared: staged into LDS once per block
// via async copies, double-buffered against the compute of the current step.
// grid = 1024 blocks; tiles = B*H*(L/16) = 8192.
// ---------------------------------------------------------------------------
__global__ __launch_bounds__(256) void attn_kernel(
    const _Float16* __restrict__ Qh, const _Float16* __restrict__ Kh,
    const _Float16* __restrict__ Vt, float* __restrict__ out) {
  __shared__ alignas(16) _Float16 kt[2][32][72];   // K tile [key][dh], padded
  __shared__ alignas(16) _Float16 vt[2][64][40];   // V^T tile [dh][key], padded
  __shared__ alignas(16) _Float16 plds[8][16][40]; // per-wave P tile

  const int tid  = threadIdx.x;
  const int wave = tid >> 5;
  const int lane = tid & 31;
  const int m16  = lane & 15;
  const int half = lane >> 4;

  const int tileId = blockIdx.x * 8 + wave;  // 0..8191
  const int qt = tileId & 127;               // q-tile in sequence
  const int hl = tileId >> 7;                // b*H + h (uniform per block)
  const int q0 = qt * 16;
  const int b  = hl >> 4;
  const int h  = hl & 15;

  const _Float16* Qb = Qh + (size_t)hl * L_ * DH_;
  const _Float16* Kb = Kh + (size_t)hl * L_ * DH_;
  const _Float16* Vb = Vt + (size_t)hl * DH_ * L_;

  // Per-thread async-copy coordinates (fixed for all iterations).
  const int krow = tid >> 3, kc = (tid & 7) * 8;   // 32 rows x 8 chunks of 16B
  const int vrow = tid >> 2, vc = (tid & 3) * 8;   // 64 rows x 4 chunks of 16B

  // Q A-fragments, held in registers for the whole key loop.
  const _Float16* qrow = Qb + (size_t)(q0 + m16) * DH_;
  v16h aq0 = frag_a_f16(qrow, half);
  v16h aq1 = frag_a_f16(qrow + 32, half);

  float mrow[8], lrow[8];
  v8f acc[4];
#pragma unroll
  for (int r = 0; r < 8; ++r) { mrow[r] = -1e30f; lrow[r] = 0.0f; }
#pragma unroll
  for (int j = 0; j < 4; ++j) acc[j] = v8f_zero();

  // Prologue: stage tile 0 into buffer 0.
  async_cp_b128(&kt[0][krow][kc], Kb + (size_t)krow * DH_ + kc);
  async_cp_b128(&vt[0][vrow][vc], Vb + (size_t)vrow * L_ + vc);
  wait_async0();
  __syncthreads();

  for (int kb = 0; kb < L_; kb += 32) {
    const int cur = (kb >> 5) & 1;
    const int nxt = cur ^ 1;
    const bool hasNext = (kb + 32) < L_;

    // Kick off async staging of the NEXT tile (overlaps with compute below).
    if (hasNext) {
      async_cp_b128(&kt[nxt][krow][kc], Kb + (size_t)(kb + 32 + krow) * DH_ + kc);
      async_cp_b128(&vt[nxt][vrow][vc], Vb + (size_t)vrow * L_ + kb + 32 + vc);
    }

    // ---- scores: two 16x16 tiles for keys [kb,kb+15], [kb+16,kb+31]
    v8f s0 = v8f_zero(), s1 = v8f_zero();
    s0 = wmma16(aq0, frag_b_f16(&kt[cur][m16][0], half), s0);
    s0 = wmma16(aq1, frag_b_f16(&kt[cur][m16][32], half), s0);
    s1 = wmma16(aq0, frag_b_f16(&kt[cur][16 + m16][0], half), s1);
    s1 = wmma16(aq1, frag_b_f16(&kt[cur][16 + m16][32], half), s1);

    __syncthreads();  // previous iteration's P reads complete before overwrite

    // ---- online softmax update over the 32 new keys
#pragma unroll
    for (int r = 0; r < 8; ++r) {
      float v = fmaxf(s0[r], s1[r]);
#pragma unroll
      for (int off = 8; off >= 1; off >>= 1)
        v = fmaxf(v, __shfl_xor(v, off, 32));   // stays inside 16-lane half
      float newm = fmaxf(mrow[r], v);
      float sc = __expf(mrow[r] - newm);
      float e0 = __expf(s0[r] - newm);
      float e1 = __expf(s1[r] - newm);
      float rs = e0 + e1;
#pragma unroll
      for (int off = 8; off >= 1; off >>= 1)
        rs += __shfl_xor(rs, off, 32);
      lrow[r] = lrow[r] * sc + rs;
      mrow[r] = newm;
#pragma unroll
      for (int j = 0; j < 4; ++j) acc[j][r] *= sc;

      int prow = r + 8 * half;    // C-layout -> row-major P tile in LDS
      plds[wave][prow][m16]      = (_Float16)e0;
      plds[wave][prow][16 + m16] = (_Float16)e1;
    }

    __syncthreads();  // P tile ordered before A-fragment reload

    // ---- P(16x32) @ V(32x64): 4 WMMAs from the staged V^T tile
    v16h pa = frag_a_f16(&plds[wave][m16][0], half);
#pragma unroll
    for (int j = 0; j < 4; ++j) {
      v16h bv = frag_b_f16(&vt[cur][j * 16 + m16][0], half);
      acc[j] = wmma16(pa, bv, acc[j]);
    }

    // Next tile must be resident, and everyone must be done reading `cur`
    // (which becomes the write target in the following iteration).
    if (hasNext) wait_async0();
    __syncthreads();
  }

  // ---- epilogue: normalize and store f32 output (B, L, H*DH)
#pragma unroll
  for (int r = 0; r < 8; ++r) {
    float inv = 1.0f / lrow[r];
    int l = q0 + r + 8 * half;
    float* orow = out + ((size_t)b * L_ + l) * D_ + h * DH_;
#pragma unroll
    for (int j = 0; j < 4; ++j) orow[j * 16 + m16] = acc[j][r] * inv;
  }
}

// ---------------------------------------------------------------------------
extern "C" void kernel_launch(void* const* d_in, const int* in_sizes, int n_in,
                              void* d_out, int out_size, void* d_ws, size_t ws_size,
                              hipStream_t stream) {
  const float* x   = (const float*)d_in[0];
  const float* Wq  = (const float*)d_in[1];
  const float* Wkv = (const float*)d_in[2];
  float* out = (float*)d_out;

  const size_t nPerTensor = (size_t)B_ * H_ * L_ * DH_;  // 8,388,608 f16 each
  _Float16* Qh = (_Float16*)d_ws;
  _Float16* Kh = Qh + nPerTensor;
  _Float16* Vt = Kh + nPerTensor;

  proj_kernel<<<dim3(64, 48), 256, 0, stream>>>(x, Wq, Wkv, Qh, Kh, Vt);
  attn_kernel<<<dim3(1024), 256, 0, stream>>>(Qh, Kh, Vt, out);
}
